// transfusionBlock_TriAttn3_46385646797058
// MI455X (gfx1250) — matrix operand order repaired and checked by reading, not compiled, not run on hardware
//
// CDNA5 / gfx1250 implementation of transfusionBlock_TriAttn3.
// Dense matmuls -> v_wmma_f32_16x16x32_bf16 (fp32->bf16 inline, bias/gelu/resid
// fused). Channel-major activations staged through per-wave LDS tiles; token-
// major activations loaded as float4. Attention (head dim 8) = online softmax.
#include <hip/hip_runtime.h>
#include <hip/hip_bf16.h>
#include <math.h>

#define BB 8
#define CDIM 128
#define CH 64
#define HH 80
#define WW 80
#define PIX (HH * WW)          // 6400
#define HEADS 8
#define DH 8
#define STCAP 112              // 100 st tokens padded to a multiple of 16

typedef __attribute__((ext_vector_type(16))) __bf16 v16bf;
typedef __attribute__((ext_vector_type(8)))  float  v8f;

__device__ __forceinline__ void cvt8(v16bf& f, int base, float4 a, float4 b) {
  f[base + 0] = (__bf16)a.x; f[base + 1] = (__bf16)a.y;
  f[base + 2] = (__bf16)a.z; f[base + 3] = (__bf16)a.w;
  f[base + 4] = (__bf16)b.x; f[base + 5] = (__bf16)b.y;
  f[base + 6] = (__bf16)b.z; f[base + 7] = (__bf16)b.w;
}

// ---------------------------------------------------------------- elementwise

__global__ void k_concat(const float* __restrict__ x0, const float* __restrict__ x1,
                         float* __restrict__ o) {
  long long i = (long long)blockIdx.x * blockDim.x + threadIdx.x;
  long long tot = (long long)BB * CDIM * PIX;
  if (i >= tot) return;
  int p = (int)(i % PIX);
  int c = (int)(i / PIX) % CDIM;
  int b = (int)(i / ((long long)PIX * CDIM));
  o[i] = (c < CH) ? x0[((long long)b * CH + c) * PIX + p]
                  : x1[((long long)b * CH + (c - CH)) * PIX + p];
}

// out[b,c,p] = in[b,c,p] + bias[c] + sum_3x3 w[c]*bn(in)
__global__ void k_dw3(const float* __restrict__ in, long long inB,
                      const float* __restrict__ w, const float* __restrict__ bias,
                      const float* __restrict__ bnG, const float* __restrict__ bnB,
                      float* __restrict__ out, int C) {
  long long i = (long long)blockIdx.x * blockDim.x + threadIdx.x;
  long long tot = (long long)BB * C * PIX;
  if (i >= tot) return;
  int p = (int)(i % PIX);
  int c = (int)(i / PIX) % C;
  int b = (int)(i / ((long long)PIX * C));
  int h = p / WW, x = p % WW;
  const float* src = in + (long long)b * inB + (long long)c * PIX;
  float s = 1.f, t = 0.f;
  if (bnG) { s = bnG[c] * rsqrtf(1.f + 1e-5f); t = bnB[c]; }
  float acc = bias[c];
#pragma unroll
  for (int dy = 0; dy < 3; ++dy)
#pragma unroll
    for (int dx = 0; dx < 3; ++dx) {
      int hh = h + dy - 1, xx = x + dx - 1;
      if (hh >= 0 && hh < HH && xx >= 0 && xx < WW) {
        float v = src[hh * WW + xx];
        acc += (v * s + t) * w[c * 9 + dy * 3 + dx];
      }
    }
  out[((long long)b * C + c) * PIX + p] = src[p] + acc;
}

__global__ void k_bn(const float* __restrict__ in, const float* __restrict__ g,
                     const float* __restrict__ bb, float* __restrict__ out, int C) {
  long long i = (long long)blockIdx.x * blockDim.x + threadIdx.x;
  long long tot = (long long)BB * C * PIX;
  if (i >= tot) return;
  int c = (int)(i / PIX) % C;
  out[i] = in[i] * (g[c] * rsqrtf(1.f + 1e-5f)) + bb[c];
}

// -------------------------------------------------------------- triplet gates
__global__ void k_pool(const float* __restrict__ X, float* __restrict__ pool,
                       int mode, int D1, int D2, int R) {
  long long i = (long long)blockIdx.x * blockDim.x + threadIdx.x;
  long long tot = (long long)BB * D1 * D2;
  if (i >= tot) return;
  int d2 = (int)(i % D2);
  int d1 = (int)(i / D2) % D1;
  int b  = (int)(i / ((long long)D1 * D2));
  float mx = -3.0e38f, sm = 0.f;
  for (int r = 0; r < R; ++r) {
    long long idx;
    if (mode == 0)      idx = (((long long)b * CDIM + d1) * HH + r)  * WW + d2;
    else if (mode == 1) idx = (((long long)b * CDIM + d2) * HH + d1) * WW + r;
    else                idx = (((long long)b * CDIM + r)  * HH + d1) * WW + d2;
    float v = X[idx];
    mx = fmaxf(mx, v);
    sm += v;
  }
  pool[(((long long)b * 2 + 0) * D1 + d1) * D2 + d2] = mx;
  pool[(((long long)b * 2 + 1) * D1 + d1) * D2 + d2] = sm / (float)R;
}

__global__ void k_gate7(const float* __restrict__ pool, const float* __restrict__ w,
                        const float* __restrict__ gs, const float* __restrict__ gb,
                        float* __restrict__ gate, int D1, int D2) {
  long long i = (long long)blockIdx.x * blockDim.x + threadIdx.x;
  long long tot = (long long)BB * D1 * D2;
  if (i >= tot) return;
  int d2 = (int)(i % D2);
  int d1 = (int)(i / D2) % D1;
  int b  = (int)(i / ((long long)D1 * D2));
  float acc = 0.f;
  for (int c2 = 0; c2 < 2; ++c2)
    for (int ky = 0; ky < 7; ++ky) {
      int y = d1 + ky - 3;
      if (y < 0 || y >= D1) continue;
      for (int kx = 0; kx < 7; ++kx) {
        int xx = d2 + kx - 3;
        if (xx < 0 || xx >= D2) continue;
        acc += pool[(((long long)b * 2 + c2) * D1 + y) * D2 + xx] * w[(c2 * 7 + ky) * 7 + kx];
      }
    }
  float z = acc * (gs[0] * rsqrtf(1.f + 1e-5f)) + gb[0];
  gate[i] = 1.f / (1.f + __expf(-z));
}

__global__ void k_tri(const float* __restrict__ x, const float* __restrict__ gcw,
                      const float* __restrict__ ghc, const float* __restrict__ ghw,
                      float* __restrict__ o) {
  long long i = (long long)blockIdx.x * blockDim.x + threadIdx.x;
  long long tot = (long long)BB * CDIM * PIX;
  if (i >= tot) return;
  int p = (int)(i % PIX);
  int c = (int)(i / PIX) % CDIM;
  int b = (int)(i / ((long long)PIX * CDIM));
  int h = p / WW, wx = p % WW;
  float gsum = gcw[((long long)b * CDIM + c) * WW + wx]
             + ghc[((long long)b * HH + h) * CDIM + c]
             + ghw[((long long)b * HH + h) * WW + wx];
  o[i] = x[i] * gsum * (1.f / 3.f);
}

// ------------------------------------------------------------------ LN (C=64)
__global__ void k_ln(const float* __restrict__ in, long long inB,
                     float* __restrict__ out, long long outB,
                     const float* __restrict__ g, const float* __restrict__ bv, int Ntok) {
  long long i = (long long)blockIdx.x * blockDim.x + threadIdx.x;
  if (i >= (long long)BB * Ntok) return;
  int t = (int)(i % Ntok);
  int b = (int)(i / Ntok);
  const float* r = in + b * inB + (long long)t * CH;
  float m = 0.f;
#pragma unroll
  for (int c = 0; c < CH; ++c) m += r[c];
  m *= (1.f / CH);
  float v = 0.f;
#pragma unroll
  for (int c = 0; c < CH; ++c) { float d = r[c] - m; v += d * d; }
  v *= (1.f / CH);
  float inv = rsqrtf(v + 1e-5f);
  float* o = out + b * outB + (long long)t * CH;
#pragma unroll
  for (int c = 0; c < CH; ++c) o[c] = (r[c] - m) * inv * g[c] + bv[c];
}

// ------------------------------------------------------------- WMMA bf16 GEMM
// C[b,m,n] = sum_k A[b,m,k]*W[n,k] + bias (+gelu) (+resid).  M multiple of 16.
// amode 0: token-major A (aSk==1, row stride aSm)  -> float4 direct loads
// amode 1: channel-major A (aSm==1, k stride aSk)  -> per-wave LDS staging
__global__ void k_gemm(const float* __restrict__ A, long long aB,
                       const float* __restrict__ W, const float* __restrict__ bias,
                       float* __restrict__ C, long long cB,
                       const float* __restrict__ R, long long rB,
                       int M, int N, int K,
                       int aSm, int aSk, int cSm, int cSn, int fuse, int amode) {
  __shared__ float tileMem[8 * 512];          // 16KB: one 16x32 f32 tile / wave
  const int lane = threadIdx.x & 31;
  const int wave = threadIdx.x >> 5;
  const int tm = (blockIdx.x * (blockDim.x >> 5) + wave) * 16;
  const int tn = blockIdx.y * 16;
  if (tm >= M) return;                        // wave-uniform: EXEC all-1s
  const int b = blockIdx.z;
  const float* Ab = A + (long long)b * aB;
  const int kh = lane >> 4;
  const int ml = lane & 15;
  const int mrow = tm + ml;
  const int ncol = tn + ml;
  const int kb = kh * 16;
  float* tile = tileMem + wave * 512;
  v8f acc = {0.f, 0.f, 0.f, 0.f, 0.f, 0.f, 0.f, 0.f};
  for (int k0 = 0; k0 < K; k0 += 32) {
    v16bf fa, fb;
    if (amode == 0) {
      const float* ar = Ab + (long long)mrow * aSm + k0 + kh * 8;
      if (k0 + 32 < K) __builtin_prefetch(ar + 32, 0, 0);   // global_prefetch_b8
      cvt8(fa, 0, *(const float4*)(ar),      *(const float4*)(ar + 4));
      cvt8(fa, 8, *(const float4*)(ar + 16), *(const float4*)(ar + 20));
    } else {
      const float* gsrc = Ab + (long long)(k0 + lane) * aSk + tm;  // 16 floats, contiguous in m
      if (k0 + 32 < K) __builtin_prefetch(gsrc + 32LL * aSk, 0, 0);
      float4 r0 = *(const float4*)(gsrc);
      float4 r1 = *(const float4*)(gsrc + 4);
      float4 r2 = *(const float4*)(gsrc + 8);
      float4 r3 = *(const float4*)(gsrc + 12);
      float4* tl = (float4*)(tile + lane * 16);                    // ds_store_b128 x4
      tl[0] = r0; tl[1] = r1; tl[2] = r2; tl[3] = r3;
      asm volatile("" ::: "memory");          // same-wave DS is in-order on CDNA5
#pragma unroll
      for (int j = 0; j < 16; ++j) {
        int kk = j + 8 * ((j >> 3) + kh);     // frag K pattern within the tile
        fa[j] = (__bf16)tile[kk * 16 + ml];
      }
      asm volatile("" ::: "memory");
    }
    const float* wr = W + (long long)ncol * K + k0 + kh * 8;       // k 8h..8h+7, 16+8h..16+8h+7
    cvt8(fb, 0, *(const float4*)(wr),      *(const float4*)(wr + 4));
    cvt8(fb, 8, *(const float4*)(wr + 16), *(const float4*)(wr + 20));
    acc = __builtin_amdgcn_wmma_f32_16x16x32_bf16(false, fa, false, fb,
                                                  (short)0, acc, false, false);
  }
  const int n = ncol;
  const int mb = tm + kh * 8;
  float bvv = bias ? bias[n] : 0.f;
  float* Cb = C + (long long)b * cB;
  const float* Rb = R ? R + (long long)b * rB : nullptr;
  float r[8];
#pragma unroll
  for (int gg = 0; gg < 8; ++gg) {
    float v = acc[gg] + bvv;
    if (fuse & 1) v = 0.5f * v * (1.f + erff(v * 0.7071067811865476f));
    r[gg] = v;
  }
  if (cSm == 1) {                              // contiguous in m: float4 stores
    float* dst = Cb + mb + (long long)n * cSn;
    if (Rb) {
      const float* rs = Rb + mb + (long long)n * cSn;
      float4 q0 = *(const float4*)rs, q1 = *(const float4*)(rs + 4);
      r[0] += q0.x; r[1] += q0.y; r[2] += q0.z; r[3] += q0.w;
      r[4] += q1.x; r[5] += q1.y; r[6] += q1.z; r[7] += q1.w;
    }
    float4 o0 = {r[0], r[1], r[2], r[3]}, o1 = {r[4], r[5], r[6], r[7]};
    *(float4*)dst = o0; *(float4*)(dst + 4) = o1;
  } else {
#pragma unroll
    for (int gg = 0; gg < 8; ++gg) {
      long long off = (long long)(mb + gg) * cSm + (long long)n * cSn;
      float v = r[gg];
      if (Rb) v += Rb[off];
      Cb[off] = v;
    }
  }
}

// --------------------------------------------------- online-softmax attention
__global__ void k_attn(const float* __restrict__ QQ, long long qB,
                       const float* __restrict__ KV, long long kvB,
                       float* __restrict__ O, long long oB,
                       const float* __restrict__ resid, long long rB,
                       int Nq, int Nk) {
  long long i = (long long)blockIdx.x * blockDim.x + threadIdx.x;
  long long tot = (long long)BB * HEADS * Nq;
  if (i >= tot) return;
  int q = (int)(i % Nq);
  int h = (int)(i / Nq) % HEADS;
  int b = (int)(i / ((long long)Nq * HEADS));
  const float scale = 0.35355339059327373f;    // 8^-0.5
  const float* qr = QQ + b * qB + (long long)q * 192 + h * DH;
  float4 q0 = *(const float4*)qr, q1 = *(const float4*)(qr + 4);
  q0.x *= scale; q0.y *= scale; q0.z *= scale; q0.w *= scale;
  q1.x *= scale; q1.y *= scale; q1.z *= scale; q1.w *= scale;
  const float* kvb = KV + b * kvB;
  float mx = -3.0e38f, l = 0.f;
  float a0 = 0.f, a1 = 0.f, a2 = 0.f, a3 = 0.f, a4 = 0.f, a5 = 0.f, a6 = 0.f, a7 = 0.f;
  for (int m = 0; m < Nk; ++m) {
    const float* kr = kvb + (long long)m * 192 + 64 + h * DH;
    float4 k0 = *(const float4*)kr, k1 = *(const float4*)(kr + 4);
    float s = q0.x * k0.x + q0.y * k0.y + q0.z * k0.z + q0.w * k0.w
            + q1.x * k1.x + q1.y * k1.y + q1.z * k1.z + q1.w * k1.w;
    float mn = fmaxf(mx, s);
    float corr = __expf(mx - mn);
    float e = __expf(s - mn);
    l = l * corr + e;
    const float* vr = kvb + (long long)m * 192 + 128 + h * DH;
    float4 v0 = *(const float4*)vr, v1 = *(const float4*)(vr + 4);
    a0 = a0 * corr + e * v0.x; a1 = a1 * corr + e * v0.y;
    a2 = a2 * corr + e * v0.z; a3 = a3 * corr + e * v0.w;
    a4 = a4 * corr + e * v1.x; a5 = a5 * corr + e * v1.y;
    a6 = a6 * corr + e * v1.z; a7 = a7 * corr + e * v1.w;
    mx = mn;
  }
  float invl = 1.f / l;
  float* orow = O + b * oB + (long long)q * CH + h * DH;
  float o[8] = {a0 * invl, a1 * invl, a2 * invl, a3 * invl,
                a4 * invl, a5 * invl, a6 * invl, a7 * invl};
  if (resid) {
    const float* rr = resid + b * rB + (long long)q * CH + h * DH;
    float4 r0 = *(const float4*)rr, r1 = *(const float4*)(rr + 4);
    o[0] += r0.x; o[1] += r0.y; o[2] += r0.z; o[3] += r0.w;
    o[4] += r1.x; o[5] += r1.y; o[6] += r1.z; o[7] += r1.w;
  }
  float4 w0 = {o[0], o[1], o[2], o[3]}, w1 = {o[4], o[5], o[6], o[7]};
  *(float4*)orow = w0; *(float4*)(orow + 4) = w1;
}

// -------------------------------------------------------------------- resize
__global__ void k_resize(const float* __restrict__ in, long long inB,
                         float* __restrict__ out, int C, int Hi, int Wi, int Ho, int Wo) {
  long long i = (long long)blockIdx.x * blockDim.x + threadIdx.x;
  long long tot = (long long)BB * C * Ho * Wo;
  if (i >= tot) return;
  int x = (int)(i % Wo);
  int y = (int)(i / Wo) % Ho;
  int c = (int)(i / ((long long)Wo * Ho)) % C;
  int b = (int)(i / ((long long)Wo * Ho * C));
  float fy = fminf(fmaxf((y + 0.5f) * ((float)Hi / Ho) - 0.5f, 0.f), (float)(Hi - 1));
  float fx = fminf(fmaxf((x + 0.5f) * ((float)Wi / Wo) - 0.5f, 0.f), (float)(Wi - 1));
  int y0 = (int)fy, x0 = (int)fx;
  int y1 = min(y0 + 1, Hi - 1), x1 = min(x0 + 1, Wi - 1);
  float wy = fy - y0, wx = fx - x0;
  const float* s = in + (long long)b * inB + (long long)c * Hi * Wi;
  float t00 = s[y0 * Wi + x0], t01 = s[y0 * Wi + x1];
  float t10 = s[y1 * Wi + x0], t11 = s[y1 * Wi + x1];
  out[i] = (t00 * (1 - wx) + t01 * wx) * (1 - wy) + (t10 * (1 - wx) + t11 * wx) * wy;
}

__global__ void k_c2t(const float* __restrict__ img, float* __restrict__ tok,
                      int C, int NT, long long tokB) {
  long long i = (long long)blockIdx.x * blockDim.x + threadIdx.x;
  long long tot = (long long)BB * C * NT;
  if (i >= tot) return;
  int p = (int)(i % NT);
  int c = (int)(i / NT) % C;
  int b = (int)(i / ((long long)NT * C));
  tok[b * tokB + (long long)p * C + c] = img[i];
}

__global__ void k_t2c(const float* __restrict__ tok, long long tokB,
                      float* __restrict__ img, int C, int NT) {
  long long i = (long long)blockIdx.x * blockDim.x + threadIdx.x;
  long long tot = (long long)BB * C * NT;
  if (i >= tot) return;
  int p = (int)(i % NT);
  int c = (int)(i / NT) % C;
  int b = (int)(i / ((long long)NT * C));
  img[i] = tok[b * tokB + (long long)p * C + c];
}

__global__ void k_cat3(const float* __restrict__ a, const float* __restrict__ bx,
                       const float* __restrict__ cx, float* __restrict__ o) {
  long long i = (long long)blockIdx.x * blockDim.x + threadIdx.x;
  long long tot = (long long)BB * 192 * PIX;
  if (i >= tot) return;
  int p = (int)(i % PIX);
  int k = (int)(i / PIX) % 192;
  int b = (int)(i / ((long long)PIX * 192));
  const float* src;
  if (k < 64)       src = a  + ((long long)b * CH + k)         * PIX;
  else if (k < 128) src = bx + ((long long)b * CH + (k - 64))  * PIX;
  else              src = cx + ((long long)b * CH + (k - 128)) * PIX;
  o[i] = src[p];
}

__global__ void k_split(const float* __restrict__ x, float* __restrict__ out) {
  long long i = (long long)blockIdx.x * blockDim.x + threadIdx.x;
  long long tot = (long long)BB * CDIM * PIX;
  if (i >= tot) return;
  int p = (int)(i % PIX);
  int c = (int)(i / PIX) % CDIM;
  int b = (int)(i / ((long long)PIX * CDIM));
  long long half = (long long)BB * CH * PIX;
  float v = x[i];
  if (c < CH) out[((long long)b * CH + c) * PIX + p] = v;
  else        out[half + ((long long)b * CH + (c - CH)) * PIX + p] = v;
}

// ------------------------------------------------------------------- drivers
static inline dim3 grid1(long long n) { return dim3((unsigned)((n + 255) / 256)); }

static inline void launch_gemm(hipStream_t st, const float* A, long long aB,
                               const float* W, const float* bias, float* C, long long cB,
                               const float* R, long long rB, int M, int N, int K,
                               int aSm, int aSk, int cSm, int cSn, int fuse, int amode) {
  int mt = (M + 15) / 16;
  dim3 g((mt + 7) / 8, N / 16, BB), b(256);
  k_gemm<<<g, b, 0, st>>>(A, aB, W, bias, C, cB, R, rB, M, N, K, aSm, aSk, cSm, cSn, fuse, amode);
}

static void run_triplet(hipStream_t stream, const float* xb, float* pool,
                        float* gcw, float* ghc, float* ghw, float* out,
                        const float* const* prm, int base) {
  long long nCW = (long long)BB * CDIM * WW;
  long long nHC = (long long)BB * HH * CDIM;
  long long nHW = (long long)BB * HH * WW;
  k_pool <<<grid1(nCW), 256, 0, stream>>>(xb, pool, 0, CDIM, WW, HH);
  k_gate7<<<grid1(nCW), 256, 0, stream>>>(pool, prm[base + 0], prm[base + 1], prm[base + 2], gcw, CDIM, WW);
  k_pool <<<grid1(nHC), 256, 0, stream>>>(xb, pool, 1, HH, CDIM, WW);
  k_gate7<<<grid1(nHC), 256, 0, stream>>>(pool, prm[base + 3], prm[base + 4], prm[base + 5], ghc, HH, CDIM);
  k_pool <<<grid1(nHW), 256, 0, stream>>>(xb, pool, 2, HH, WW, CDIM);
  k_gate7<<<grid1(nHW), 256, 0, stream>>>(pool, prm[base + 6], prm[base + 7], prm[base + 8], ghw, HH, WW);
  long long nAll = (long long)BB * CDIM * PIX;
  k_tri<<<grid1(nAll), 256, 0, stream>>>(xb, gcw, ghc, ghw, out);
}

extern "C" void kernel_launch(void* const* d_in, const int* in_sizes, int n_in,
                              void* d_out, int out_size, void* d_ws, size_t ws_size,
                              hipStream_t stream) {
  (void)in_sizes; (void)out_size; (void)ws_size;
  const float* prm[128];
  for (int i = 0; i < n_in && i < 128; ++i) prm[i] = (const float*)d_in[i];
  const float* x0 = prm[0];
  const float* x1 = prm[1];

  float* wsf = (float*)d_ws;
  size_t off = 0;
  auto alloc = [&](size_t n) -> float* {
    float* p = wsf + off;
    off += (n + 63) & ~(size_t)63;
    return p;
  };
  const size_t PBIG = (size_t)BB * CDIM * PIX;
  const size_t PHLF = (size_t)BB * CH * PIX;
  float* XA = alloc(PBIG); float* XB = alloc(PBIG); float* XC = alloc(PBIG);
  float* TB = alloc(PBIG); float* TT = alloc(PBIG);
  float* POOL = alloc((size_t)BB * 2 * CDIM * WW);
  float* GCW = alloc((size_t)BB * CDIM * WW);
  float* GHC = alloc((size_t)BB * HH * CDIM);
  float* GHW = alloc((size_t)BB * HH * WW);
  float* RGB = alloc(PHLF);  float* STAB = alloc(PHLF); float* BNR = alloc(PHLF);
  float* RGBO = alloc(PHLF); float* IRF = alloc(PHLF);  float* STF = alloc(PHLF);
  float* IMG40 = alloc((size_t)BB * CH * 1600);
  float* IMG10 = alloc((size_t)BB * CH * 100);
  float* IRTOK = alloc((size_t)BB * 1600 * CH);
  float* IRT2  = alloc((size_t)BB * 1600 * CH);
  float* IRB2  = alloc((size_t)BB * 1600 * CH);
  float* LNB   = alloc((size_t)BB * 1600 * CH);
  float* QKV1  = alloc((size_t)BB * 1600 * 192);
  float* QKVI2 = alloc((size_t)BB * 1600 * 192);
  float* STT   = alloc((size_t)BB * STCAP * CH);
  float* STT2  = alloc((size_t)BB * STCAP * CH);
  float* STX   = alloc((size_t)BB * STCAP * CH);
  float* STY   = alloc((size_t)BB * STCAP * CH);
  float* QKVS1 = alloc((size_t)BB * STCAP * 192);
  float* QKVS2 = alloc((size_t)BB * STCAP * 192);
  float* QKVR  = alloc((size_t)BB * PIX * 192);
  float* ATTR  = alloc((size_t)BB * PIX * CH);
  float* ATTI  = alloc((size_t)BB * 1600 * CH);
  float* ATTS1 = alloc((size_t)BB * STCAP * CH);
  float* ATTS2 = alloc((size_t)BB * STCAP * CH);
  float* CAT   = alloc((size_t)BB * 192 * PIX);
  float* HML   = alloc((size_t)BB * 512 * PIX);

  long long nBIG = (long long)PBIG, nHALF = (long long)PHLF;

  // x = concat(x0,x1); x = x + dwconv3(x)        (XB == x == residual)
  k_concat<<<grid1(nBIG), 256, 0, stream>>>(x0, x1, XA);
  k_dw3<<<grid1(nBIG), 256, 0, stream>>>(XA, (long long)CDIM * PIX, prm[2], prm[3],
                                         nullptr, nullptr, XB, CDIM);
  // three (bn -> triplet -> 1x1 conv) blocks
  k_bn<<<grid1(nBIG), 256, 0, stream>>>(XB, prm[4], prm[5], TB, CDIM);
  run_triplet(stream, TB, POOL, GCW, GHC, GHW, TT, prm, 58);
  launch_gemm(stream, TT, (long long)CDIM * PIX, prm[16], prm[17], XA, (long long)CDIM * PIX,
              nullptr, 0, PIX, CDIM, CDIM, 1, PIX, 1, PIX, 0, 1);
  k_bn<<<grid1(nBIG), 256, 0, stream>>>(XA, prm[6], prm[7], TB, CDIM);
  run_triplet(stream, TB, POOL, GCW, GHC, GHW, TT, prm, 67);
  launch_gemm(stream, TT, (long long)CDIM * PIX, prm[18], prm[19], XC, (long long)CDIM * PIX,
              nullptr, 0, PIX, CDIM, CDIM, 1, PIX, 1, PIX, 0, 1);
  k_bn<<<grid1(nBIG), 256, 0, stream>>>(XC, prm[8], prm[9], TB, CDIM);
  run_triplet(stream, TB, POOL, GCW, GHC, GHW, TT, prm, 76);
  launch_gemm(stream, TT, (long long)CDIM * PIX, prm[20], prm[21], XA, (long long)CDIM * PIX,
              nullptr, 0, PIX, CDIM, CDIM, 1, PIX, 1, PIX, 0, 1);
  // rgb = rgb + dwconv3(bn(rgb, nc1))
  k_dw3<<<grid1(nHALF), 256, 0, stream>>>(XA + (long long)CH * PIX, (long long)CDIM * PIX,
                                          prm[24], prm[25], prm[12], prm[13], RGB, CH);
  // ir: resize 80->40, tokens, LN, qkv proj, self-attn (+resid)
  k_resize<<<grid1((long long)BB * CH * 1600), 256, 0, stream>>>(XA, (long long)CDIM * PIX,
                                                                 IMG40, CH, HH, WW, 40, 40);
  k_c2t<<<grid1((long long)BB * CH * 1600), 256, 0, stream>>>(IMG40, IRTOK, CH, 1600,
                                                              (long long)1600 * CH);
  k_ln<<<grid1((long long)BB * 1600), 256, 0, stream>>>(IRTOK, (long long)1600 * CH,
                                                        LNB, (long long)1600 * CH,
                                                        prm[28], prm[29], 1600);
  launch_gemm(stream, LNB, (long long)1600 * CH, prm[36], prm[37], QKV1, (long long)1600 * 192,
              nullptr, 0, 1600, 192, CH, CH, 1, 192, 1, 0, 0);
  k_attn<<<grid1((long long)BB * HEADS * 1600), 256, 0, stream>>>(
      QKV1, (long long)1600 * 192, QKV1, (long long)1600 * 192,
      IRT2, (long long)1600 * CH, IRTOK, (long long)1600 * CH, 1600, 1600);
  // stable = conv1x1(x, cd); st: resize 80->10, tokens (padded to 112 rows)
  launch_gemm(stream, XA, (long long)CDIM * PIX, prm[26], prm[27], STAB, (long long)CH * PIX,
              nullptr, 0, PIX, CH, CDIM, 1, PIX, 1, PIX, 0, 1);
  k_resize<<<grid1((long long)BB * CH * 100), 256, 0, stream>>>(STAB, (long long)CH * PIX,
                                                                IMG10, CH, HH, WW, 10, 10);
  k_c2t<<<grid1((long long)BB * CH * 100), 256, 0, stream>>>(IMG10, STT, CH, 100,
                                                             (long long)STCAP * CH);
  k_ln<<<grid1((long long)BB * 100), 256, 0, stream>>>(STT, (long long)STCAP * CH,
                                                       LNB, (long long)STCAP * CH,
                                                       prm[30], prm[31], 100);
  launch_gemm(stream, LNB, (long long)STCAP * CH, prm[38], prm[39], QKVS1, (long long)STCAP * 192,
              nullptr, 0, STCAP, 192, CH, CH, 1, 192, 1, 0, 0);
  k_attn<<<grid1((long long)BB * HEADS * 100), 256, 0, stream>>>(
      QKVS1, (long long)STCAP * 192, QKVS1, (long long)STCAP * 192,
      STT2, (long long)STCAP * CH, STT, (long long)STCAP * CH, 100, 100);
  // cross-modal qkv projections
  k_bn<<<grid1(nHALF), 256, 0, stream>>>(RGB, prm[14], prm[15], BNR, CH);
  launch_gemm(stream, BNR, (long long)CH * PIX, prm[40], prm[41], QKVR, (long long)PIX * 192,
              nullptr, 0, PIX, 192, CH, 1, PIX, 192, 1, 0, 1);
  k_ln<<<grid1((long long)BB * 1600), 256, 0, stream>>>(IRT2, (long long)1600 * CH,
                                                        LNB, (long long)1600 * CH,
                                                        prm[32], prm[33], 1600);
  launch_gemm(stream, LNB, (long long)1600 * CH, prm[42], prm[43], QKVI2, (long long)1600 * 192,
              nullptr, 0, 1600, 192, CH, CH, 1, 192, 1, 0, 0);
  k_ln<<<grid1((long long)BB * 100), 256, 0, stream>>>(STT2, (long long)STCAP * CH,
                                                       LNB, (long long)STCAP * CH,
                                                       prm[34], prm[35], 100);
  launch_gemm(stream, LNB, (long long)STCAP * CH, prm[44], prm[45], QKVS2, (long long)STCAP * 192,
              nullptr, 0, STCAP, 192, CH, CH, 1, 192, 1, 0, 0);
  // cross attentions + output projections (residual fused into GEMM epilogue)
  k_attn<<<grid1((long long)BB * HEADS * PIX), 256, 0, stream>>>(
      QKVR, (long long)PIX * 192, QKVS2, (long long)STCAP * 192,
      ATTR, (long long)PIX * CH, nullptr, 0, PIX, 100);
  launch_gemm(stream, ATTR, (long long)PIX * CH, prm[46], prm[47], RGBO, (long long)CH * PIX,
              RGB, (long long)CH * PIX, PIX, CH, CH, CH, 1, 1, PIX, 0, 0);
  k_attn<<<grid1((long long)BB * HEADS * 100), 256, 0, stream>>>(
      QKVS2, (long long)STCAP * 192, QKVR, (long long)PIX * 192,
      ATTS1, (long long)STCAP * CH, nullptr, 0, 100, PIX);
  launch_gemm(stream, ATTS1, (long long)STCAP * CH, prm[50], prm[51], STX, (long long)STCAP * CH,
              STT2, (long long)STCAP * CH, STCAP, CH, CH, CH, 1, CH, 1, 0, 0);
  k_attn<<<grid1((long long)BB * HEADS * 100), 256, 0, stream>>>(
      QKVS2, (long long)STCAP * 192, QKVI2, (long long)1600 * 192,
      ATTS2, (long long)STCAP * CH, nullptr, 0, 100, 1600);
  launch_gemm(stream, ATTS2, (long long)STCAP * CH, prm[52], prm[53], STY, (long long)STCAP * CH,
              STX, (long long)STCAP * CH, STCAP, CH, CH, CH, 1, CH, 1, 0, 0);
  k_attn<<<grid1((long long)BB * HEADS * 1600), 256, 0, stream>>>(
      QKVI2, (long long)1600 * 192, QKVS2, (long long)STCAP * 192,
      ATTI, (long long)1600 * CH, nullptr, 0, 1600, 100);
  launch_gemm(stream, ATTI, (long long)1600 * CH, prm[48], prm[49], IRB2, (long long)1600 * CH,
              IRT2, (long long)1600 * CH, 1600, CH, CH, CH, 1, CH, 1, 0, 0);
  // upsample ir/st back to 80x80, concat, c2 (+residual XB)
  k_t2c<<<grid1((long long)BB * CH * 1600), 256, 0, stream>>>(IRB2, (long long)1600 * CH,
                                                              IMG40, CH, 1600);
  k_resize<<<grid1(nHALF), 256, 0, stream>>>(IMG40, (long long)CH * 1600, IRF, CH, 40, 40, HH, WW);
  k_t2c<<<grid1((long long)BB * CH * 100), 256, 0, stream>>>(STY, (long long)STCAP * CH,
                                                             IMG10, CH, 100);
  k_resize<<<grid1(nHALF), 256, 0, stream>>>(IMG10, (long long)CH * 100, STF, CH, 10, 10, HH, WW);
  k_cat3<<<grid1((long long)BB * 192 * PIX), 256, 0, stream>>>(RGBO, IRF, STF, CAT);
  launch_gemm(stream, CAT, (long long)192 * PIX, prm[22], prm[23], XC, (long long)CDIM * PIX,
              XB, (long long)CDIM * PIX, PIX, CDIM, 192, 1, PIX, 1, PIX, 0, 1);
  // output block: bn -> triplet -> m1(gelu) -> m2(+resid)
  k_bn<<<grid1(nBIG), 256, 0, stream>>>(XC, prm[10], prm[11], TB, CDIM);
  run_triplet(stream, TB, POOL, GCW, GHC, GHW, TT, prm, 85);
  launch_gemm(stream, TT, (long long)CDIM * PIX, prm[54], prm[55], HML, (long long)512 * PIX,
              nullptr, 0, PIX, 512, CDIM, 1, PIX, 1, PIX, 1, 1);
  launch_gemm(stream, HML, (long long)512 * PIX, prm[56], prm[57], XA, (long long)CDIM * PIX,
              XC, (long long)CDIM * PIX, PIX, CDIM, 512, 1, PIX, 1, PIX, 0, 1);
  k_split<<<grid1(nBIG), 256, 0, stream>>>(XA, (float*)d_out);
}
